// OrthogonalWindowAttention_15281493639413
// MI455X (gfx1250) — matrix operand order repaired and checked by reading, not compiled, not run on hardware
//
#include <hip/hip_runtime.h>

typedef __attribute__((ext_vector_type(16))) _Float16 v16h;
typedef __attribute__((ext_vector_type(8)))  _Float16 v8h;
typedef __attribute__((ext_vector_type(4)))  _Float16 v4h;
typedef __attribute__((ext_vector_type(8)))  float    v8f;

#define NWIN 4096
#define NTOK 49      // tokens per window
#define CDIM 128
#define HEADS 4
#define HD 32

// LDS strides (in elements), padded to dodge bank conflicts; all row strides
// are multiples of 8 halves (16B) so v8h loads stay aligned.
#define XS 136   // staged X / pre-proj Z (aliased buffer), f16
#define YS 520   // Y = [q|k|v|q_basis], f16
#define SS 68    // S / P scratch, f32
#define AS 72    // A1 / B2 softmax matrices, f16
#define ZS 136   // pre-proj combined output stride (same buffer as Xs)

__device__ __forceinline__ v16h cat8(v8h lo, v8h hi) {
  return __builtin_shufflevector(lo, hi, 0,1,2,3,4,5,6,7,8,9,10,11,12,13,14,15);
}

// A-operand (16x32 f16): lane m = lane&15; koff = 8*(lane>>4);
// elements 0..15 -> K = 2*koff..2*koff+15 (contiguous, matches B operands)
__device__ __forceinline__ v16h load_a(const _Float16* row, int koff) {
  return cat8(*(const v8h*)(row + 2 * koff), *(const v8h*)(row + 2 * koff + 8));
}

// ---------------------------------------------------------------------------
// Prep: transpose+convert weights to f16 (B-operand friendly: row = out-dim,
// contiguous K), fold the 1/sqrt(hd) scale into the q columns, build the
// combined 512-wide bias vector.
// ---------------------------------------------------------------------------
__global__ void owa_prep(const float* __restrict__ qkv_w, const float* __restrict__ qkv_b,
                         const float* __restrict__ qb_w, const float* __restrict__ qb_b,
                         const float* __restrict__ proj_w,
                         _Float16* __restrict__ Wt, _Float16* __restrict__ Pt,
                         float* __restrict__ cb) {
  int idx = blockIdx.x * blockDim.x + threadIdx.x;   // 0 .. 512*128-1
  if (idx >= 512 * 128) return;
  int j = idx >> 7, k = idx & 127;
  const float scale = 0.17677669529663687f;          // hd^-0.5, hd = 32
  float w = (j < 3 * CDIM) ? qkv_w[k * (3 * CDIM) + j] : qb_w[k * CDIM + (j - 3 * CDIM)];
  if (j < CDIM) w *= scale;                          // q columns pre-scaled
  Wt[j * CDIM + k] = (_Float16)w;
  if (j < CDIM) Pt[j * CDIM + k] = (_Float16)proj_w[k * CDIM + j];
  if (k == 0) {
    float bv = (j < 3 * CDIM) ? qkv_b[j] : qb_b[j - 3 * CDIM];
    if (j < CDIM) bv *= scale;
    cb[j] = bv;
  }
}

// ---------------------------------------------------------------------------
// Fused orthogonal window attention: one workgroup (8 wave32 waves) per window.
// ---------------------------------------------------------------------------
__global__ __launch_bounds__(256) void owa_main(
    const float* __restrict__ x, const int* __restrict__ rel_index,
    const float* __restrict__ bias_table, const float* __restrict__ proj_b,
    const float* __restrict__ sel, const _Float16* __restrict__ Wt,
    const _Float16* __restrict__ Pt, const float* __restrict__ cb,
    float* __restrict__ out) {
  // XZ is time-multiplexed: X staging (through GEMM1) then Z (pre-proj out).
  __shared__ __attribute__((aligned(32))) _Float16 XZ[64 * XS];
  __shared__ __attribute__((aligned(32))) _Float16 Yb[64 * YS];
  __shared__ __attribute__((aligned(16))) float    Sb[64 * SS];
  __shared__ __attribute__((aligned(16))) float    Pb[64 * SS];
  __shared__ __attribute__((aligned(32))) _Float16 A1f[64 * AS];
  __shared__ __attribute__((aligned(32))) _Float16 B2f[64 * AS];
  __shared__ float rmax_s[64], rsum_s[64], cmax_s[64], csum_s[64];
  __shared__ float vks[64], s2s[64], vinv[64], qbinv[64];
  // 4-way partial reductions (indexed [g*64 + row/col])
  __shared__ float rpm[256], rps[256], cpm[256], cps[256], vkp[256], s2p[256];

  const int b    = blockIdx.x;
  const int tid  = threadIdx.x;
  const int wave = tid >> 5;
  const int lane = tid & 31;
  const int kh   = lane >> 4;      // which K-half this lane carries
  const int l15  = lane & 15;
  const int koff = kh * 8;

  // ---- stage X: 49x128 fp32 -> f16 in LDS, rows 49..63 zeroed ----
  const float* xb = x + (size_t)b * (NTOK * CDIM);
  for (int i = tid; i < (64 * CDIM) / 4; i += 256) {
    int n = i >> 5;
    int c0 = (i & 31) << 2;
    v4h hv;
    if (n < NTOK) {
      const float4 f = *(const float4*)(xb + n * CDIM + c0);
      hv = v4h{(_Float16)f.x, (_Float16)f.y, (_Float16)f.z, (_Float16)f.w};
    } else {
      hv = v4h{(_Float16)0.f, (_Float16)0.f, (_Float16)0.f, (_Float16)0.f};
    }
    *(v4h*)(&XZ[n * XS + c0]) = hv;
  }
  __syncthreads();

  // ---- GEMM1: Y(64x512) = X @ [qkv_w | q_basis_w] + bias (WMMA f16->f32) ----
  for (int t = wave; t < 128; t += 8) {          // 4x32 tiles of 16x16
    const int ti = t >> 5, tj = t & 31;
    v8f acc = {};
    const _Float16* arow = &XZ[(16 * ti + l15) * XS];
    const _Float16* brow = &Wt[(16 * tj + l15) * CDIM + kh * 16];
#pragma unroll
    for (int kk = 0; kk < 4; ++kk) {
      v16h a = load_a(arow + 32 * kk, koff);
      v16h bm = *(const v16h*)(brow + 32 * kk);  // 16 contiguous K per lane
      acc = __builtin_amdgcn_wmma_f32_16x16x32_f16(false, a, false, bm, (short)0, acc, false, false);
    }
    const int j = 16 * tj + l15;
    const float bj = cb[j];
#pragma unroll
    for (int r = 0; r < 8; ++r) {
      int n = 16 * ti + r + 8 * kh;
      Yb[n * YS + j] = (_Float16)(acc[r] + bj);
    }
  }
  __syncthreads();

  const float cw = 1.f / (1.f + __expf(-sel[0]));

  for (int h = 0; h < HEADS; ++h) {
    const int qo = h * HD, ko = CDIM + h * HD, vo = 2 * CDIM + h * HD, qbo = 3 * CDIM + h * HD;

    // ---- 1/||v||^2 and 1/||q_basis||^2 (waves 0-3; waves 4-7 start S/P) ----
    if (tid < 64) {
      int m = tid; float s = 0.f;
      if (m < NTOK) {
        for (int d = 0; d < HD; ++d) { float vv = (float)Yb[m * YS + vo + d]; s += vv * vv; }
        vinv[m] = 1.f / s;
      } else vinv[m] = 0.f;
    } else if (tid < 128) {
      int n = tid - 64; float s = 0.f;
      if (n < NTOK) {
        for (int d = 0; d < HD; ++d) { float qv = (float)Yb[n * YS + qbo + d]; s += qv * qv; }
        qbinv[n] = 1.f / s;
      } else qbinv[n] = 0.f;
    }

    // ---- S = q_hat k^T + bias (16 tiles) and P = q_basis v^T (16 tiles) ----
    for (int t = wave; t < 32; t += 8) {
      const bool isS = (t < 16);
      const int tt = t & 15, ti = tt >> 2, tj = tt & 3;
      const int ac = isS ? qo : qbo;
      const int bc = isS ? ko : vo;
      const _Float16* arow = &Yb[(16 * ti + l15) * YS + ac];
      v16h a = load_a(arow, koff);
      const _Float16* brow = &Yb[(16 * tj + l15) * YS + bc + kh * 16];
      v16h bm = cat8(*(const v8h*)(brow), *(const v8h*)(brow + 8));
      v8f acc = {};
      acc = __builtin_amdgcn_wmma_f32_16x16x32_f16(false, a, false, bm, (short)0, acc, false, false);
      float* obuf = isS ? Sb : Pb;
#pragma unroll
      for (int r = 0; r < 8; ++r) {
        int n = 16 * ti + r + 8 * kh, m = 16 * tj + l15;
        float val = acc[r];
        if (isS && n < NTOK && m < NTOK)
          val += bias_table[rel_index[n * NTOK + m] * HEADS + h];
        obuf[n * SS + m] = val;
      }
    }
    __syncthreads();

    // ---- softmax stats, 4-way parallel partials over all 256 threads ----
    {
      const int nn = tid & 63, g = tid >> 6, base = g * 16;
      float mx = -1e30f, s = 0.f;        // row nn, cols base..base+15
      float cmx = -1e30f, cs = 0.f;      // col nn, rows base..base+15
      if (nn < NTOK) {
        for (int i = 0; i < 16; ++i) {
          int m = base + i;
          if (m < NTOK) mx = fmaxf(mx, Sb[nn * SS + m]);
        }
        for (int i = 0; i < 16; ++i) {
          int m = base + i;
          if (m < NTOK) s += __expf(Sb[nn * SS + m] - mx);
        }
        for (int i = 0; i < 16; ++i) {
          int n2 = base + i;
          if (n2 < NTOK) cmx = fmaxf(cmx, Sb[n2 * SS + nn]);
        }
        for (int i = 0; i < 16; ++i) {
          int n2 = base + i;
          if (n2 < NTOK) cs += __expf(Sb[n2 * SS + nn] - cmx);
        }
      }
      rpm[g * 64 + nn] = mx;  rps[g * 64 + nn] = s;
      cpm[g * 64 + nn] = cmx; cps[g * 64 + nn] = cs;
    }
    __syncthreads();
    // combine partials (rescaled-sum trick: S = sum_g s_g * exp(m_g - M))
    if (tid < 64) {
      int n = tid;
      float mx = fmaxf(fmaxf(rpm[n], rpm[64 + n]), fmaxf(rpm[128 + n], rpm[192 + n]));
      float s = rps[n] * __expf(rpm[n] - mx) + rps[64 + n] * __expf(rpm[64 + n] - mx)
              + rps[128 + n] * __expf(rpm[128 + n] - mx) + rps[192 + n] * __expf(rpm[192 + n] - mx);
      rmax_s[n] = mx; rsum_s[n] = (n < NTOK) ? 1.f / s : 0.f;
    } else if (tid < 128) {
      int m = tid - 64;
      float mx = fmaxf(fmaxf(cpm[m], cpm[64 + m]), fmaxf(cpm[128 + m], cpm[192 + m]));
      float s = cps[m] * __expf(cpm[m] - mx) + cps[64 + m] * __expf(cpm[64 + m] - mx)
              + cps[128 + m] * __expf(cpm[128 + m] - mx) + cps[192 + m] * __expf(cpm[192 + m] - mx);
      cmax_s[m] = mx; csum_s[m] = (m < NTOK) ? 1.f / s : 0.f;
    }
    __syncthreads();

    // ---- materialize A1 = softmax_keys(S), B2 = softmax_queries(S)*P/||v||^2,
    //      with 4-way partial row reductions vk = sum A1*P and s2 = sum A2 ----
    {
      const int n = tid & 63, g = tid >> 6, base = g * 16;
      float vkacc = 0.f, s2acc = 0.f;
      if (n < NTOK) {
        float rm = rmax_s[n], ri = rsum_s[n];
        for (int i = 0; i < 16; ++i) {
          int m = base + i;
          _Float16 a1h = (_Float16)0.f, b2h = (_Float16)0.f;
          if (m < NTOK) {
            float sv = Sb[n * SS + m], pv = Pb[n * SS + m];
            float a1 = __expf(sv - rm) * ri;
            float a2 = __expf(sv - cmax_s[m]) * csum_s[m];
            a1h = (_Float16)a1;
            b2h = (_Float16)(a2 * pv * vinv[m]);
            vkacc += a1 * pv;
            s2acc += a2;
          }
          A1f[n * AS + m] = a1h;
          B2f[n * AS + m] = b2h;
        }
      } else {
        for (int i = 0; i < 16; ++i) {
          A1f[n * AS + base + i] = (_Float16)0.f;
          B2f[n * AS + base + i] = (_Float16)0.f;
        }
      }
      vkp[g * 64 + n] = vkacc; s2p[g * 64 + n] = s2acc;
    }
    __syncthreads();
    if (tid < 64) {
      vks[tid] = vkp[tid] + vkp[64 + tid] + vkp[128 + tid] + vkp[192 + tid];
    } else if (tid < 128) {
      int n = tid - 64;
      s2s[n] = s2p[n] + s2p[64 + n] + s2p[128 + n] + s2p[192 + n];
    }
    __syncthreads();

    // ---- M1 = A1 @ v, M2 = B2 @ v, orthogonal epilogue -> Z (1 tile/wave) ----
    {
      const int ti = wave >> 1, tj2 = wave & 1;
      v8f acc1 = {}, acc2 = {};
#pragma unroll
      for (int kk = 0; kk < 2; ++kk) {
        v16h bm;
        const int mbase = 32 * kk + kh * 16;
        const int dcol = vo + 16 * tj2 + l15;
#pragma unroll
        for (int e = 0; e < 16; ++e) bm[e] = Yb[(mbase + e) * YS + dcol];
        const _Float16* a1row = &A1f[(16 * ti + l15) * AS + 32 * kk];
        v16h a1 = load_a(a1row, koff);
        acc1 = __builtin_amdgcn_wmma_f32_16x16x32_f16(false, a1, false, bm, (short)0, acc1, false, false);
        const _Float16* a2row = &B2f[(16 * ti + l15) * AS + 32 * kk];
        v16h a2 = load_a(a2row, koff);
        acc2 = __builtin_amdgcn_wmma_f32_16x16x32_f16(false, a2, false, bm, (short)0, acc2, false, false);
      }
#pragma unroll
      for (int r = 0; r < 8; ++r) {
        int n = 16 * ti + r + 8 * kh;
        int d = 16 * tj2 + l15;
        float qbv = (float)Yb[n * YS + qbo + d];
        float vor = acc1[r] - vks[n] * qbv * qbinv[n];     // v_orths
        float qor = qbv * s2s[n] - acc2[r];                // q_orths
        XZ[n * ZS + h * HD + d] = (_Float16)(cw * vor + (1.f - cw) * qor);
      }
    }
    __syncthreads();
  }

  // ---- proj: out(49x128) = Z @ proj_w + proj_b ----
  float* ob = out + (size_t)b * (NTOK * CDIM);
  for (int t = wave; t < 32; t += 8) {                 // 4x8 tiles
    const int ti = t >> 3, tj = t & 7;
    v8f acc = {};
    const _Float16* arow = &XZ[(16 * ti + l15) * ZS];
    const _Float16* brow = &Pt[(16 * tj + l15) * CDIM + kh * 16];
#pragma unroll
    for (int kk = 0; kk < 4; ++kk) {
      v16h a = load_a(arow + 32 * kk, koff);
      v16h bm = *(const v16h*)(brow + 32 * kk);
      acc = __builtin_amdgcn_wmma_f32_16x16x32_f16(false, a, false, bm, (short)0, acc, false, false);
    }
    const int j = 16 * tj + l15;
    const float pbj = proj_b[j];
#pragma unroll
    for (int r = 0; r < 8; ++r) {
      int n = 16 * ti + r + 8 * kh;
      if (n < NTOK) ob[n * CDIM + j] = acc[r] + pbj;
    }
  }
}

extern "C" void kernel_launch(void* const* d_in, const int* in_sizes, int n_in,
                              void* d_out, int out_size, void* d_ws, size_t ws_size,
                              hipStream_t stream) {
  const float* x          = (const float*)d_in[0];
  const float* qkv_w      = (const float*)d_in[1];
  const float* qkv_b      = (const float*)d_in[2];
  const float* qb_w       = (const float*)d_in[3];
  const float* qb_b       = (const float*)d_in[4];
  const float* proj_w     = (const float*)d_in[5];
  const float* proj_b     = (const float*)d_in[6];
  const float* bias_table = (const float*)d_in[7];
  const float* sel        = (const float*)d_in[8];
  const int*   rel_index  = (const int*)d_in[9];
  float* out = (float*)d_out;

  char* ws = (char*)d_ws;
  _Float16* Wt = (_Float16*)ws;                                  // 512x128 f16
  _Float16* Pt = (_Float16*)(ws + 512 * 128 * 2);                // 128x128 f16
  float*    cb = (float*)(ws + 512 * 128 * 2 + 128 * 128 * 2);   // 512 f32

  owa_prep<<<256, 256, 0, stream>>>(qkv_w, qkv_b, qb_w, qb_b, proj_w, Wt, Pt, cb);
  owa_main<<<NWIN, 256, 0, stream>>>(x, rel_index, bias_table, proj_b, sel, Wt, Pt, cb, out);
}
